// CausalSelfAttention_6665789244057
// MI455X (gfx1250) — matrix operand order repaired and checked by reading, not compiled, not run on hardware
//
#include <hip/hip_runtime.h>

typedef __bf16 bf16;
typedef __attribute__((ext_vector_type(16))) __bf16 v16bf;
typedef __attribute__((ext_vector_type(8)))  __bf16 v8bf;
typedef __attribute__((ext_vector_type(8)))  float  v8f;

#define WMMA_BF16(a, b, c) \
  __builtin_amdgcn_wmma_f32_16x16x32_bf16(false, (a), false, (b), (short)0, (c), false, false)

union Frag { v16bf v; v8bf h[2]; };

// B=4 S=2048 D=1024 H=16 hd=64
static constexpr int Bsz = 4, Sq = 2048, Dm = 1024, Hn = 16, Hd = 64;

// ---------------------------------------------------------------- f32 -> bf16
__global__ void cvt_f32_bf16(const float* __restrict__ in, bf16* __restrict__ out, int n) {
  int i = blockIdx.x * blockDim.x + threadIdx.x;
  if (i < n) out[i] = (bf16)in[i];
}

// ---------------------------------------------------------------- QKV GEMM
// C[8192,3072] = X[8192,1024] @ W[1024,3072]; double-buffered LDS pipeline,
// epilogue scatters q/k/v (part/head are block-uniform: one scalar branch).
__global__ __launch_bounds__(128) void gemm_qkv(
    const bf16* __restrict__ X, const bf16* __restrict__ W,
    bf16* __restrict__ Q, bf16* __restrict__ K, bf16* __restrict__ Vt) {
  __shared__ __align__(16) bf16 lA[2][64 * 32];   // [m][k]
  __shared__ __align__(16) bf16 lB[2][64 * 32];   // transposed [n][k]
  const int tid = threadIdx.x;
  const int lane = tid & 31, wave = tid >> 5;
  const int laneN = lane & 15, laneHi = lane >> 4;
  const int m0 = blockIdx.y * 64;
  const int n0 = blockIdx.x * 64;
  const int NW = 3 * Dm;

  const int am = tid >> 1, akh = (tid & 1) * 16;
  const int bk = tid >> 2, bns = (tid & 3) * 16;
  const bf16* aptr = X + (size_t)(m0 + am) * Dm + akh;
  const bf16* bptr = W + (size_t)bk * NW + n0 + bns;

  v8bf ra0, ra1, rw0, rw1;
  auto gload = [&](int k0) {
    const v8bf* sa = (const v8bf*)(aptr + k0);
    ra0 = sa[0]; ra1 = sa[1];
    const v8bf* sb = (const v8bf*)(bptr + (size_t)k0 * NW);
    rw0 = sb[0]; rw1 = sb[1];
  };
  auto lstore = [&](int buf) {
    *(v8bf*)&lA[buf][am * 32 + akh]     = ra0;
    *(v8bf*)&lA[buf][am * 32 + akh + 8] = ra1;
#pragma unroll
    for (int j = 0; j < 8; ++j) {
      lB[buf][(bns + j) * 32 + bk]     = rw0[j];
      lB[buf][(bns + 8 + j) * 32 + bk] = rw1[j];
    }
  };

  v8f acc[4] = {};
  gload(0);
  lstore(0);
  __syncthreads();
  for (int kt = 0; kt < 32; ++kt) {
    const int cur = kt & 1;
    if (kt < 31) gload((kt + 1) * 32);
    Frag af;  // A: row M = lane&15, K = (e&7)+(e>=8?16:0)+hi*8
    int ar = wave * 16 + laneN;
    af.h[0] = *(const v8bf*)&lA[cur][ar * 32 + laneHi * 8];
    af.h[1] = *(const v8bf*)&lA[cur][ar * 32 + 16 + laneHi * 8];
#pragma unroll
    for (int nt = 0; nt < 4; ++nt) {
      Frag bf_;  // B: col N = lane&15, K = (e&7)+(e>=8?16:0)+hi*8
      int bn = nt * 16 + laneN;
      bf_.h[0] = *(const v8bf*)&lB[cur][bn * 32 + laneHi * 8];
      bf_.h[1] = *(const v8bf*)&lB[cur][bn * 32 + 16 + laneHi * 8];
      acc[nt] = WMMA_BF16(af.v, bf_.v, acc[nt]);
    }
    if (kt < 31) lstore(cur ^ 1);
    __syncthreads();
  }

  // epilogue: part and head are uniform over the block (n0 % 64 == 0, hd == 64)
  const int part = n0 >> 10;
  const int h = (n0 & 1023) >> 6;
  if (part < 2) {
    bf16* dst = (part == 0) ? Q : K;
#pragma unroll
    for (int nt = 0; nt < 4; ++nt)
#pragma unroll
      for (int v = 0; v < 8; ++v) {
        int m = m0 + wave * 16 + v + laneHi * 8;
        int b = m >> 11, s = m & 2047;
        dst[(((size_t)(b * Hn + h)) * Sq + s) * Hd + nt * 16 + laneN] = (bf16)acc[nt][v];
      }
  } else {
#pragma unroll
    for (int nt = 0; nt < 4; ++nt)
#pragma unroll
      for (int v = 0; v < 8; ++v) {
        int m = m0 + wave * 16 + v + laneHi * 8;
        int b = m >> 11, s = m & 2047;
        Vt[(((size_t)(b * Hn + h)) * Hd + nt * 16 + laneN) * Sq + s] = (bf16)acc[nt][v];
      }
  }
}

// ---------------------------------------------------------------- RoPE in place
__global__ void rope_kernel(bf16* __restrict__ Q, bf16* __restrict__ K) {
  int i = blockIdx.x * blockDim.x + threadIdx.x;  // B*H*S*32
  int dd = i & 31;
  int s  = (i >> 5) & (Sq - 1);
  int bh = i >> 16;
  size_t base = ((size_t)bh * Sq + s) * Hd;
  float inv = __expf(-(float)dd * 0.28782313662425580f);  // 10000^(-dd/32)
  float th = (float)s * inv;
  float c = __cosf(th), sn = __sinf(th);
  float q1 = (float)Q[base + dd], q2 = (float)Q[base + dd + 32];
  Q[base + dd]      = (bf16)(q1 * c - q2 * sn);
  Q[base + dd + 32] = (bf16)(q2 * c + q1 * sn);
  float k1 = (float)K[base + dd], k2 = (float)K[base + dd + 32];
  K[base + dd]      = (bf16)(k1 * c - k2 * sn);
  K[base + dd + 32] = (bf16)(k2 * c + k1 * sn);
}

// ---------------------------------------------------------------- flash attention
// grid (S/64, B*H), 4 independent waves; each: 16 queries x hd=64, online softmax.
__global__ __launch_bounds__(128) void attn_flash(
    const bf16* __restrict__ Q, const bf16* __restrict__ K,
    const bf16* __restrict__ Vt, bf16* __restrict__ O) {
  __shared__ __align__(16) bf16 lP[4][16 * 32];  // per-wave P tile [q][key]
  const int tid = threadIdx.x;
  const int lane = tid & 31, wave = tid >> 5;
  const int laneN = lane & 15, laneHi = lane >> 4;
  const int bh = blockIdx.y;
  const int qw = blockIdx.x * 64 + wave * 16;
  const bf16* Qp = Q + (size_t)bh * Sq * Hd;
  const bf16* Kp = K + (size_t)bh * Sq * Hd;
  const bf16* Vp = Vt + (size_t)bh * Hd * Sq;

  Frag qf[2];
  int qrow = qw + laneN;
#pragma unroll
  for (int ks = 0; ks < 2; ++ks) {
    qf[ks].h[0] = *(const v8bf*)(Qp + (size_t)qrow * Hd + ks * 32 + laneHi * 8);
    qf[ks].h[1] = *(const v8bf*)(Qp + (size_t)qrow * Hd + ks * 32 + 16 + laneHi * 8);
  }
  v8f o[4] = {};
  float mrow[8], lrow[8];
#pragma unroll
  for (int v = 0; v < 8; ++v) { mrow[v] = -3.0e38f; lrow[v] = 0.f; }
  const float scale = 0.125f;  // hd^-0.5
  bf16* P = &lP[wave][0];

  int nkb = qw / 32 + 1;
  for (int kb = 0; kb < nkb; ++kb) {
    int k0 = kb * 32;
    if (kb + 1 < nkb) {  // prefetch next key block (K rows / V^T rows)
      __builtin_prefetch(Kp + (size_t)(k0 + 32 + lane) * Hd, 0, 3);
      __builtin_prefetch(Vp + (size_t)(2 * lane) * Sq + k0 + 32, 0, 3);
      __builtin_prefetch(Vp + (size_t)(2 * lane + 1) * Sq + k0 + 32, 0, 3);
    }
    // scores: S[16q x 32k] = Q @ K^T, B-fragments straight from global
    v8f sacc[2] = {};
#pragma unroll
    for (int nt = 0; nt < 2; ++nt) {
      int key = k0 + nt * 16 + laneN;
#pragma unroll
      for (int ks = 0; ks < 2; ++ks) {
        Frag kf;
        kf.h[0] = *(const v8bf*)(Kp + (size_t)key * Hd + ks * 32 + laneHi * 8);
        kf.h[1] = *(const v8bf*)(Kp + (size_t)key * Hd + ks * 32 + 16 + laneHi * 8);
        sacc[nt] = WMMA_BF16(qf[ks].v, kf.v, sacc[nt]);
      }
    }
    // scale + causal mask + running row max
    float rmax[8];
#pragma unroll
    for (int v = 0; v < 8; ++v) rmax[v] = -3.0e38f;
#pragma unroll
    for (int nt = 0; nt < 2; ++nt)
#pragma unroll
      for (int v = 0; v < 8; ++v) {
        int q = qw + v + laneHi * 8;
        int key = k0 + nt * 16 + laneN;
        float s = sacc[nt][v] * scale;
        s = (key <= q) ? s : -3.0e38f;
        sacc[nt][v] = s;
        rmax[v] = fmaxf(rmax[v], s);
      }
#pragma unroll
    for (int v = 0; v < 8; ++v) {
      float x = rmax[v];
#pragma unroll
      for (int msk = 1; msk <= 8; msk <<= 1) x = fmaxf(x, __shfl_xor(x, msk, 32));
      float mnew = fmaxf(mrow[v], x);
      rmax[v] = __expf(mrow[v] - mnew);  // alpha
      mrow[v] = mnew;
    }
    // P = exp(s - m), row sums, C-layout -> LDS [q][key]
    float rsum[8] = {};
#pragma unroll
    for (int nt = 0; nt < 2; ++nt)
#pragma unroll
      for (int v = 0; v < 8; ++v) {
        float p = __expf(sacc[nt][v] - mrow[v]);
        rsum[v] += p;
        P[(v + laneHi * 8) * 32 + nt * 16 + laneN] = (bf16)p;
      }
#pragma unroll
    for (int v = 0; v < 8; ++v) {
      float x = rsum[v];
#pragma unroll
      for (int msk = 1; msk <= 8; msk <<= 1) x += __shfl_xor(x, msk, 32);
      lrow[v] = lrow[v] * rmax[v] + x;
    }
#pragma unroll
    for (int nt = 0; nt < 4; ++nt)
#pragma unroll
      for (int v = 0; v < 8; ++v) o[nt][v] *= rmax[v];
    // same-wave LDS is in-order; fence blocks compiler reordering, wait drains stores
    asm volatile("s_wait_dscnt 0" ::: "memory");
    // reload P as A fragment, multiply by V^T B fragments from global
    Frag pf;
    pf.h[0] = *(const v8bf*)&P[laneN * 32 + laneHi * 8];
    pf.h[1] = *(const v8bf*)&P[laneN * 32 + 16 + laneHi * 8];
#pragma unroll
    for (int nt = 0; nt < 4; ++nt) {
      Frag vf;
      int hdc = nt * 16 + laneN;
      vf.h[0] = *(const v8bf*)(Vp + (size_t)hdc * Sq + k0 + laneHi * 8);
      vf.h[1] = *(const v8bf*)(Vp + (size_t)hdc * Sq + k0 + 16 + laneHi * 8);
      o[nt] = WMMA_BF16(pf.v, vf.v, o[nt]);
    }
    asm volatile("s_wait_dscnt 0" ::: "memory");
  }
  // finalize: O[b,s,D] bf16
  const int b = bh >> 4, h = bh & 15;
#pragma unroll
  for (int nt = 0; nt < 4; ++nt)
#pragma unroll
    for (int v = 0; v < 8; ++v) {
      int q = qw + v + laneHi * 8;
      float val = o[nt][v] / lrow[v];
      size_t tok = (size_t)b * Sq + q;
      O[tok * Dm + h * Hd + nt * 16 + laneN] = (bf16)val;
    }
}

// ---------------------------------------------------------------- out GEMM
// out[8192,1024] (f32) = O[8192,1024] @ Wout[1024,1024], double-buffered.
__global__ __launch_bounds__(128) void gemm_out(
    const bf16* __restrict__ A, const bf16* __restrict__ W, float* __restrict__ out) {
  __shared__ __align__(16) bf16 lA[2][64 * 32];
  __shared__ __align__(16) bf16 lB[2][64 * 32];
  const int tid = threadIdx.x;
  const int lane = tid & 31, wave = tid >> 5;
  const int laneN = lane & 15, laneHi = lane >> 4;
  const int m0 = blockIdx.y * 64;
  const int n0 = blockIdx.x * 64;
  const int NW = Dm;

  const int am = tid >> 1, akh = (tid & 1) * 16;
  const int bk = tid >> 2, bns = (tid & 3) * 16;
  const bf16* aptr = A + (size_t)(m0 + am) * Dm + akh;
  const bf16* bptr = W + (size_t)bk * NW + n0 + bns;

  v8bf ra0, ra1, rw0, rw1;
  auto gload = [&](int k0) {
    const v8bf* sa = (const v8bf*)(aptr + k0);
    ra0 = sa[0]; ra1 = sa[1];
    const v8bf* sb = (const v8bf*)(bptr + (size_t)k0 * NW);
    rw0 = sb[0]; rw1 = sb[1];
  };
  auto lstore = [&](int buf) {
    *(v8bf*)&lA[buf][am * 32 + akh]     = ra0;
    *(v8bf*)&lA[buf][am * 32 + akh + 8] = ra1;
#pragma unroll
    for (int j = 0; j < 8; ++j) {
      lB[buf][(bns + j) * 32 + bk]     = rw0[j];
      lB[buf][(bns + 8 + j) * 32 + bk] = rw1[j];
    }
  };

  v8f acc[4] = {};
  gload(0);
  lstore(0);
  __syncthreads();
  for (int kt = 0; kt < 32; ++kt) {
    const int cur = kt & 1;
    if (kt < 31) gload((kt + 1) * 32);
    Frag af;
    int ar = wave * 16 + laneN;
    af.h[0] = *(const v8bf*)&lA[cur][ar * 32 + laneHi * 8];
    af.h[1] = *(const v8bf*)&lA[cur][ar * 32 + 16 + laneHi * 8];
#pragma unroll
    for (int nt = 0; nt < 4; ++nt) {
      Frag bf_;
      int bn = nt * 16 + laneN;
      bf_.h[0] = *(const v8bf*)&lB[cur][bn * 32 + laneHi * 8];
      bf_.h[1] = *(const v8bf*)&lB[cur][bn * 32 + 16 + laneHi * 8];
      acc[nt] = WMMA_BF16(af.v, bf_.v, acc[nt]);
    }
    if (kt < 31) lstore(cur ^ 1);
    __syncthreads();
  }
#pragma unroll
  for (int nt = 0; nt < 4; ++nt)
#pragma unroll
    for (int v = 0; v < 8; ++v) {
      int m = m0 + wave * 16 + v + laneHi * 8;
      int n = n0 + nt * 16 + laneN;
      out[(size_t)m * Dm + n] = acc[nt][v];
    }
}

// ---------------------------------------------------------------- launcher
extern "C" void kernel_launch(void* const* d_in, const int* in_sizes, int n_in,
                              void* d_out, int out_size, void* d_ws, size_t ws_size,
                              hipStream_t stream) {
  const float* x    = (const float*)d_in[0];  // [4,2048,1024]
  const float* Wqkv = (const float*)d_in[1];  // [1024,3072]
  const float* Wout = (const float*)d_in[2];  // [1024,1024]
  float* out = (float*)d_out;                 // [4,2048,1024]

  char* ws = (char*)d_ws;
  constexpr size_t NTOK = (size_t)Bsz * Sq;            // 8192
  constexpr size_t SZ_QKV = NTOK * Dm * sizeof(bf16);  // 16 MB each
  bf16* Qb  = (bf16*)(ws);
  bf16* Kb  = (bf16*)(ws + SZ_QKV);
  bf16* Vt  = (bf16*)(ws + 2 * SZ_QKV);
  bf16* Ob  = (bf16*)(ws + 3 * SZ_QKV);
  bf16* Xb  = (bf16*)(ws + 4 * SZ_QKV);
  bf16* Wqb = (bf16*)(ws + 5 * SZ_QKV);
  bf16* Wob = (bf16*)(ws + 5 * SZ_QKV + (size_t)Dm * 3 * Dm * sizeof(bf16));

  // 1) convert inputs to bf16
  cvt_f32_bf16<<<(int)(NTOK * Dm / 256), 256, 0, stream>>>(x, Xb, (int)(NTOK * Dm));
  cvt_f32_bf16<<<(Dm * 3 * Dm) / 256, 256, 0, stream>>>(Wqkv, Wqb, Dm * 3 * Dm);
  cvt_f32_bf16<<<(Dm * Dm) / 256, 256, 0, stream>>>(Wout, Wob, Dm * Dm);

  // 2) QKV projection + scatter into attention layouts
  gemm_qkv<<<dim3(3 * Dm / 64, NTOK / 64), 128, 0, stream>>>(Xb, Wqb, Qb, Kb, Vt);

  // 3) RoPE in place on Q, K
  rope_kernel<<<(Bsz * Hn * Sq * 32) / 256, 256, 0, stream>>>(Qb, Kb);

  // 4) causal flash attention
  attn_flash<<<dim3(Sq / 64, Bsz * Hn), 128, 0, stream>>>(Qb, Kb, Vt, Ob);

  // 5) output projection -> f32
  gemm_out<<<dim3(Dm / 64, NTOK / 64), 128, 0, stream>>>(Ob, Wob, out);
}
